// VisionAttention_65996467470391
// MI455X (gfx1250) — matrix operand order repaired and checked by reading, not compiled, not run on hardware
//
#include <hip/hip_runtime.h>
#include <math.h>

#define SEQ_LEN 4096
#define EMBED   1280
#define NHEADS  16
#define HDIM    80
#define HDPAD   96
#define NSEG    4

typedef __bf16 bf16;
typedef __attribute__((ext_vector_type(8)))  __bf16 bf16x8;
typedef __attribute__((ext_vector_type(16))) __bf16 bf16x16;
typedef __attribute__((ext_vector_type(8)))  float  f32x8;

union BF16V { bf16x16 v; bf16x8 h[2]; };

__device__ __forceinline__ bf16 f2bf(float f) {
  unsigned u = __builtin_bit_cast(unsigned, f);
  unsigned r = (u + 0x7FFFu + ((u >> 16) & 1u)) >> 16;
  unsigned short s = (unsigned short)r;
  return __builtin_bit_cast(bf16, s);
}
__device__ __forceinline__ float bf2f(bf16 x) { return (float)x; }

// xor-butterfly over the 16-lane halves using DS_SWIZZLE group-of-32:
// imm = and_mask[4:0]=0x1f | or_mask[9:5]=0 | xor_mask[14:10]
#define SWZ_XOR(v, xm)                                                        \
  __builtin_bit_cast(float, __builtin_amdgcn_ds_swizzle(                      \
      __builtin_bit_cast(int, (v)), 0x1f | ((xm) << 10)))

// A operand (16xK tile, M rows in lanes). Lane L<16: row L, K=[k0..k0+7],[k0+16..k0+23]
// Lane L>=16: row L-16, K=[k0+8..k0+15],[k0+24..k0+31]
__device__ __forceinline__ bf16x16 load_a(const bf16* base, int ld, int row, int k0, int lane) {
  const bf16* p = base + (size_t)(row + (lane & 15)) * ld + k0 + ((lane >> 4) << 3);
  BF16V u;
  u.h[0] = *(const bf16x8*)(p);
  u.h[1] = *(const bf16x8*)(p + 16);
  return u.v;
}
// B operand ((K=32)x16 tile, N cols in lanes), source stored N-major, K-contiguous.
// Lane L<16: col L, K=[k0..k0+7],[k0+16..k0+23]; Lane L>=16: col L-16, K=[k0+8..k0+15],[k0+24..k0+31]
__device__ __forceinline__ bf16x16 load_b(const bf16* base, int ld, int col, int k0, int lane) {
  const bf16* p = base + (size_t)(col + (lane & 15)) * ld + k0 + ((lane >> 4) << 3);
  BF16V u;
  u.h[0] = *(const bf16x8*)(p);
  u.h[1] = *(const bf16x8*)(p + 16);
  return u.v;
}

#define WMMA_BF16(a, b, c) \
  __builtin_amdgcn_wmma_f32_16x16x32_bf16(false, (a), false, (b), (short)0, (c), false, false)

// ---------------------------------------------------------------- convert
__global__ void convert_kernel(const float* __restrict__ in, bf16* __restrict__ out, int n) {
  int i = blockIdx.x * blockDim.x + threadIdx.x;
  int stride = gridDim.x * blockDim.x;
  for (; i < n; i += stride) out[i] = f2bf(in[i]);
}

// ---------------------------------------------------------------- GEMM: C = A * B^T + bias
// Register double-buffered: prefetch k+32 operands while issuing current WMMAs.
template <bool OUT_BF16>
__global__ __launch_bounds__(256) void gemm_bt_kernel(
    const bf16* __restrict__ A, const bf16* __restrict__ B,
    const float* __restrict__ bias, float* __restrict__ Cf, bf16* __restrict__ Cb,
    int M, int N, int K)
{
  int lane = threadIdx.x & 31;
  int w    = threadIdx.x >> 5;
  int wr   = w & 1;
  int wc   = w >> 1;
  int row0 = blockIdx.x * 64 + wr * 32;
  int col0 = blockIdx.y * 256 + wc * 64;

  f32x8 acc[2][4];
#pragma unroll
  for (int i = 0; i < 2; i++)
#pragma unroll
    for (int j = 0; j < 4; j++) acc[i][j] = {};

  bf16x16 a0 = load_a(A, K, row0,      0, lane);
  bf16x16 a1 = load_a(A, K, row0 + 16, 0, lane);
  bf16x16 b0 = load_b(B, K, col0,      0, lane);
  bf16x16 b1 = load_b(B, K, col0 + 16, 0, lane);
  bf16x16 b2 = load_b(B, K, col0 + 32, 0, lane);
  bf16x16 b3 = load_b(B, K, col0 + 48, 0, lane);

  for (int k0 = 0; k0 < K; k0 += 32) {
    int kn = (k0 + 32 < K) ? (k0 + 32) : 0;  // wrap keeps addresses valid
    bf16x16 na0 = load_a(A, K, row0,      kn, lane);
    bf16x16 na1 = load_a(A, K, row0 + 16, kn, lane);
    bf16x16 nb0 = load_b(B, K, col0,      kn, lane);
    bf16x16 nb1 = load_b(B, K, col0 + 16, kn, lane);
    bf16x16 nb2 = load_b(B, K, col0 + 32, kn, lane);
    bf16x16 nb3 = load_b(B, K, col0 + 48, kn, lane);

    acc[0][0] = WMMA_BF16(a0, b0, acc[0][0]);
    acc[0][1] = WMMA_BF16(a0, b1, acc[0][1]);
    acc[0][2] = WMMA_BF16(a0, b2, acc[0][2]);
    acc[0][3] = WMMA_BF16(a0, b3, acc[0][3]);
    acc[1][0] = WMMA_BF16(a1, b0, acc[1][0]);
    acc[1][1] = WMMA_BF16(a1, b1, acc[1][1]);
    acc[1][2] = WMMA_BF16(a1, b2, acc[1][2]);
    acc[1][3] = WMMA_BF16(a1, b3, acc[1][3]);

    a0 = na0; a1 = na1; b0 = nb0; b1 = nb1; b2 = nb2; b3 = nb3;
  }

  int rbase = row0 + ((lane >> 4) << 3);
  int cbase = col0 + (lane & 15);
#pragma unroll
  for (int ni = 0; ni < 4; ni++) {
    float bv = bias[cbase + 16 * ni];
#pragma unroll
    for (int mi = 0; mi < 2; mi++)
#pragma unroll
      for (int r = 0; r < 8; r++) {
        int row = rbase + 16 * mi + r;
        int col = cbase + 16 * ni;
        float v = acc[mi][ni][r] + bv;
        if (OUT_BF16) Cb[(size_t)row * N + col] = f2bf(v);
        else          Cf[(size_t)row * N + col] = v;
      }
  }
}

// ---------------------------------------------------------------- RoPE + head-major relayout
__global__ __launch_bounds__(128) void rope_kernel(
    const bf16* __restrict__ qkv, const float* __restrict__ cosT,
    const float* __restrict__ sinT, bf16* __restrict__ Qh,
    bf16* __restrict__ Kh, bf16* __restrict__ Vt)
{
  int s = blockIdx.x;
  int h = blockIdx.y;
  int d = threadIdx.x;  // 0..95
  size_t qi = ((size_t)h * SEQ_LEN + s) * HDPAD + d;
  size_t vi = ((size_t)h * HDPAD + d) * SEQ_LEN + s;
  if (d >= HDIM) {
    Qh[qi] = f2bf(0.f); Kh[qi] = f2bf(0.f); Vt[vi] = f2bf(0.f);
    return;
  }
  const bf16* rowp = qkv + (size_t)s * (3 * EMBED);
  int base = h * HDIM;
  float q  = bf2f(rowp[base + d]);
  float k  = bf2f(rowp[EMBED + base + d]);
  float vv = bf2f(rowp[2 * EMBED + base + d]);
  int   dp  = (d < 40) ? d + 40 : d - 40;
  float sgn = (d < 40) ? -1.f : 1.f;
  float qp = sgn * bf2f(rowp[base + dp]);
  float kp = sgn * bf2f(rowp[EMBED + base + dp]);
  float c  = cosT[s * HDIM + d];
  float sn = sinT[s * HDIM + d];
  const float SCALING = 8.94427190999915878564f;  // sqrt(80); ref multiplies scores by it
  Qh[qi] = f2bf((q * c + qp * sn) * SCALING);
  Kh[qi] = f2bf(k * c + kp * sn);
  Vt[vi] = f2bf(vv);
}

// ---------------------------------------------------------------- flash attention per (head, 16-q tile)
__global__ __launch_bounds__(32) void attn_kernel(
    const bf16* __restrict__ Qh, const bf16* __restrict__ Kh,
    const bf16* __restrict__ Vt, const int* __restrict__ cu,
    bf16* __restrict__ attn_out)
{
  __shared__ __align__(16) bf16 psh[16 * 64];  // P tile: 16 q-rows x 64 keys
  int lane = threadIdx.x;
  int t = blockIdx.x, h = blockIdx.y;
  int r0 = t * 16;
  int s0 = 0, s1 = SEQ_LEN;
#pragma unroll
  for (int i = 0; i < NSEG; i++) {
    int a = cu[i], b = cu[i + 1];
    if (r0 >= a && r0 < b) { s0 = a; s1 = b; }
  }
  const bf16* Qb = Qh + (size_t)h * SEQ_LEN * HDPAD;
  const bf16* Kb = Kh + (size_t)h * SEQ_LEN * HDPAD;
  const bf16* Vb = Vt + (size_t)h * HDPAD * SEQ_LEN;

  bf16x16 aq[3];
#pragma unroll
  for (int dc = 0; dc < 3; dc++) aq[dc] = load_a(Qb, HDPAD, r0, dc * 32, lane);

  bf16x16 onesB;
  {
    bf16 one = f2bf(1.0f);
#pragma unroll
    for (int i = 0; i < 16; i++) onesB[i] = one;
  }

  f32x8 O[6];
#pragma unroll
  for (int n = 0; n < 6; n++) O[n] = {};
  float m[8], l[8];
#pragma unroll
  for (int r = 0; r < 8; r++) { m[r] = -3.0e38f; l[r] = 0.f; }

  int col = lane & 15;
  int rowbase = (lane >> 4) << 3;

  for (int kc = s0; kc < s1; kc += 64) {
    // ---- scores: four 16x16 tiles over 64 keys, K-loop = 3 x (d=32)
    f32x8 sc[4];
#pragma unroll
    for (int sub = 0; sub < 4; sub++) {
      f32x8 c = {};
#pragma unroll
      for (int dc = 0; dc < 3; dc++) {
        bf16x16 bk = load_b(Kb, HDPAD, kc + 16 * sub, dc * 32, lane);
        c = WMMA_BF16(aq[dc], bk, c);
      }
      sc[sub] = c;
    }
    bool valid[4];
#pragma unroll
    for (int sub = 0; sub < 4; sub++) valid[sub] = (kc + 16 * sub + col) < s1;

    // ---- online softmax: row max via ds_swizzle butterfly, exp, P -> LDS
    float f[8];
#pragma unroll
    for (int r = 0; r < 8; r++) {
      float v = -3.0e38f;
#pragma unroll
      for (int sub = 0; sub < 4; sub++) v = fmaxf(v, valid[sub] ? sc[sub][r] : -3.0e38f);
      v = fmaxf(v, SWZ_XOR(v, 1));
      v = fmaxf(v, SWZ_XOR(v, 2));
      v = fmaxf(v, SWZ_XOR(v, 4));
      v = fmaxf(v, SWZ_XOR(v, 8));
      float mn = fmaxf(m[r], v);
      f[r] = __expf(m[r] - mn);
      m[r] = mn;
      int row = r + rowbase;  // C layout: M = r + 8*(lane>=16)
#pragma unroll
      for (int sub = 0; sub < 4; sub++) {
        float pv = valid[sub] ? __expf(sc[sub][r] - mn) : 0.f;
        psh[row * 64 + sub * 16 + col] = f2bf(pv);
      }
    }
#pragma unroll
    for (int n = 0; n < 6; n++)
#pragma unroll
      for (int r = 0; r < 8; r++) O[n][r] *= f[r];

    __asm__ volatile("s_wait_dscnt 0" ::: "memory");  // DS in-order per wave

    // ---- P back as WMMA A operands (two 16x32 tiles)
    bf16x16 ap0 = load_a(psh, 64, 0, 0,  lane);
    bf16x16 ap1 = load_a(psh, 64, 0, 32, lane);

    // ---- row sums l += via WMMA against ones (lands exactly in per-lane l[] layout)
    f32x8 sums = {};
    sums = WMMA_BF16(ap0, onesB, sums);
    sums = WMMA_BF16(ap1, onesB, sums);
#pragma unroll
    for (int r = 0; r < 8; r++) l[r] = l[r] * f[r] + sums[r];

    // ---- P x V: Vt rows are d-major, key-contiguous
#pragma unroll
    for (int n = 0; n < 6; n++) {
      bf16x16 bv0 = load_b(Vb, SEQ_LEN, n * 16, kc,      lane);
      O[n] = WMMA_BF16(ap0, bv0, O[n]);
      bf16x16 bv1 = load_b(Vb, SEQ_LEN, n * 16, kc + 32, lane);
      O[n] = WMMA_BF16(ap1, bv1, O[n]);
    }
  }

  float inv[8];
#pragma unroll
  for (int r = 0; r < 8; r++) inv[r] = (l[r] > 0.f) ? 1.f / l[r] : 0.f;
  int rbase = r0 + rowbase;
#pragma unroll
  for (int n = 0; n < 6; n++) {
    int d = n * 16 + col;
    if (d < HDIM) {
#pragma unroll
      for (int r = 0; r < 8; r++) {
        attn_out[(size_t)(rbase + r) * EMBED + h * HDIM + d] = f2bf(O[n][r] * inv[r]);
      }
    }
  }
}

// ---------------------------------------------------------------- launcher
extern "C" void kernel_launch(void* const* d_in, const int* in_sizes, int n_in,
                              void* d_out, int out_size, void* d_ws, size_t ws_size,
                              hipStream_t stream)
{
  const float* hidden = (const float*)d_in[0];
  const float* w_qkv  = (const float*)d_in[1];
  const float* b_qkv  = (const float*)d_in[2];
  const float* w_proj = (const float*)d_in[3];
  const float* b_proj = (const float*)d_in[4];
  const float* cosT   = (const float*)d_in[5];
  const float* sinT   = (const float*)d_in[6];
  const int*   cu     = (const int*)d_in[7];
  float* out = (float*)d_out;

  char* p = (char*)d_ws;
  auto carve = [&](size_t elems) -> bf16* {
    bf16* r = (bf16*)p;
    size_t bytes = (elems * sizeof(bf16) + 255) & ~(size_t)255;
    p += bytes;
    return r;
  };
  bf16* hs_b    = carve((size_t)SEQ_LEN * EMBED);
  bf16* wqkv_b  = carve((size_t)3 * EMBED * EMBED);
  bf16* wproj_b = carve((size_t)EMBED * EMBED);
  bf16* qkv_raw = carve((size_t)SEQ_LEN * 3 * EMBED);
  bf16* Qh      = carve((size_t)NHEADS * SEQ_LEN * HDPAD);
  bf16* Kh      = carve((size_t)NHEADS * SEQ_LEN * HDPAD);
  bf16* Vt      = carve((size_t)NHEADS * HDPAD * SEQ_LEN);
  bf16* attn_b  = carve((size_t)SEQ_LEN * EMBED);

  convert_kernel<<<256, 256, 0, stream>>>(hidden, hs_b,    SEQ_LEN * EMBED);
  convert_kernel<<<256, 256, 0, stream>>>(w_qkv,  wqkv_b,  3 * EMBED * EMBED);
  convert_kernel<<<256, 256, 0, stream>>>(w_proj, wproj_b, EMBED * EMBED);

  dim3 g1(SEQ_LEN / 64, (3 * EMBED) / 256);
  gemm_bt_kernel<true><<<g1, 256, 0, stream>>>(hs_b, wqkv_b, b_qkv, nullptr, qkv_raw,
                                               SEQ_LEN, 3 * EMBED, EMBED);

  rope_kernel<<<dim3(SEQ_LEN, NHEADS), HDPAD, 0, stream>>>(qkv_raw, cosT, sinT, Qh, Kh, Vt);

  attn_kernel<<<dim3(SEQ_LEN / 16, NHEADS), 32, 0, stream>>>(Qh, Kh, Vt, cu, attn_b);

  dim3 g2(SEQ_LEN / 64, EMBED / 256);
  gemm_bt_kernel<false><<<g2, 256, 0, stream>>>(attn_b, wproj_b, b_proj, out, nullptr,
                                                SEQ_LEN, EMBED, EMBED);
}
